// MaskGuidedVisionTransformer_13391708029713
// MI455X (gfx1250) — matrix operand order repaired and checked by reading, not compiled, not run on hardware
//
#include <hip/hip_runtime.h>
#include <hip/hip_bf16.h>

// ---------------------------------------------------------------------------
// MaskGuidedVisionTransformer forward for MI455X (gfx1250, wave32, WMMA).
// Heavy GEMMs: v_wmma_f32_16x16x32_bf16, f32 accumulate, branchless clamped
// vector loads (b128) with on-the-fly f32->bf16 conversion. K is always a
// multiple of 32 (attention token axis padded to 224). Weights stay f32 in
// the 192MB L2 and are reused across all 3152 activation rows.
// ---------------------------------------------------------------------------

#define BATCH  16
#define NTOK   197
#define NPAD   224        // padded token axis for attention prob / V^T (K dim)
#define NPAT   196
#define DIMC   768
#define NHEAD  12
#define HDIM   64
#define NLAYER 12
#define MLPD   3072
#define GRD    14
#define BN     (BATCH*NTOK)   // 3152
#define BNP    (BATCH*NPAT)   // 3136

typedef __attribute__((ext_vector_type(16))) __bf16 bf16x16;
typedef __attribute__((ext_vector_type(8)))  float  floatx8;
typedef __attribute__((ext_vector_type(4), aligned(4))) float float4u; // 16B load, 4B align ok
typedef long long ll;

__device__ __forceinline__ float gelu_f(float x){ return 0.5f*x*(1.0f + erff(x*0.7071067811865475f)); }
__device__ __forceinline__ float sigmoid_f(float x){ return 1.0f/(1.0f+expf(-x)); }
__device__ __forceinline__ float softplus_f(float x){ return fmaxf(x,0.f) + log1pf(expf(-fabsf(x))); }
__device__ __forceinline__ float act_apply(float x, int a){
  if (a==1) return gelu_f(x);
  if (a==2) return tanhf(x);
  if (a==3) return sigmoid_f(x);
  if (a==4) return softplus_f(x);
  return x;
}
__device__ __forceinline__ int imin(int a, int b){ return a<b?a:b; }

// ---------------- WMMA fragment loaders (wave32 layouts per ISA 7.12.2) ----
// A 16x32 bf16: lanes 0-15 row M=lane: elems0..7=K0..7, elems8..15=K16..23
//               lanes16-31 row M=lane-16: elems0..7=K8..15, elems8..15=K24..31
// rowBase is pre-offset with + hf*8 ; chunks at [k0..k0+7] and [k0+16..k0+23].
__device__ __forceinline__ bf16x16 frag_a(const float* __restrict__ rowBase, int k0)
{
  float4u x0 = *(const float4u*)(rowBase + k0);
  float4u x1 = *(const float4u*)(rowBase + k0 + 4);
  float4u x2 = *(const float4u*)(rowBase + k0 + 16);
  float4u x3 = *(const float4u*)(rowBase + k0 + 20);
  bf16x16 r;
#pragma unroll
  for (int j=0;j<4;++j){
    r[j]    = (__bf16)x0[j];
    r[4+j]  = (__bf16)x1[j];
    r[8+j]  = (__bf16)x2[j];
    r[12+j] = (__bf16)x3[j];
  }
  return r;
}
// B 32x16 bf16 (KxN), element (k,n)=W[n,k]: same K order as A.
// rowBase pre-offset with + hf*8 ; chunks at [k0..k0+7] and [k0+16..k0+23].
__device__ __forceinline__ bf16x16 frag_b(const float* __restrict__ rowBase, int k0)
{
  float4u x0 = *(const float4u*)(rowBase + k0);
  float4u x1 = *(const float4u*)(rowBase + k0 + 4);
  float4u x2 = *(const float4u*)(rowBase + k0 + 16);
  float4u x3 = *(const float4u*)(rowBase + k0 + 20);
  bf16x16 r;
#pragma unroll
  for (int j=0;j<4;++j){
    r[j]    = (__bf16)x0[j];
    r[4+j]  = (__bf16)x1[j];
    r[8+j]  = (__bf16)x2[j];
    r[12+j] = (__bf16)x3[j];
  }
  return r;
}

// ---------------- generic batched NT GEMM: C = epi(A @ W^T + bias) + R -----
// 128 threads = 4 waves; wave computes a 32x32 tile (2x2 WMMA frags).
// K must be a multiple of 32. Out-of-range rows/cols are handled by clamping
// the load row (garbage lands only in C rows/cols that are masked at store).
__global__ __launch_bounds__(128) void wmma_gemm_nt(
    const float* __restrict__ Abase, ll sA1, ll sA2,
    const float* __restrict__ Wbase, ll sW1, ll sW2,
    const float* __restrict__ bias,
    float* __restrict__ Cbase, ll sC1, ll sC2,
    const float* __restrict__ Rbase,
    int M, int Nout, int K, int lda, int ldw, int ldc,
    int nb2, int epi)
{
  int lane = threadIdx.x & 31;
  int wv   = threadIdx.x >> 5;
  int z    = blockIdx.z;
  int b1 = z / nb2, b2 = z % nb2;
  const float* A  = Abase + b1*sA1 + b2*sA2;
  const float* Wp = Wbase + b1*sW1 + b2*sW2;
  float*       C  = Cbase + b1*sC1 + b2*sC2;
  const float* R  = Rbase ? (Rbase + b1*sC1 + b2*sC2) : (const float*)0;
  int wm0 = blockIdx.y*64 + (wv>>1)*32;
  int wn0 = blockIdx.x*64 + (wv&1)*32;
  if (wm0 >= M || wn0 >= Nout) return;   // wave-uniform: EXEC all-1s at WMMA

  int nl = lane & 15, hf = lane >> 4;
  // clamped row indices -> unconditional loads, masked stores
  int mA0 = imin(wm0      + nl, M-1);
  int mA1 = imin(wm0 + 16 + nl, M-1);
  int nB0 = imin(wn0      + nl, Nout-1);
  int nB1 = imin(wn0 + 16 + nl, Nout-1);
  const float* aP0 = A  + (size_t)mA0*lda + hf*8;
  const float* aP1 = A  + (size_t)mA1*lda + hf*8;
  const float* bP0 = Wp + (size_t)nB0*ldw + hf*8;
  const float* bP1 = Wp + (size_t)nB1*ldw + hf*8;

  floatx8 acc[2][2] = {};
  for (int k0 = 0; k0 < K; k0 += 32) {
    bf16x16 a0 = frag_a(aP0, k0);
    bf16x16 a1 = frag_a(aP1, k0);
    bf16x16 b0 = frag_b(bP0, k0);
    bf16x16 b1 = frag_b(bP1, k0);
    acc[0][0] = __builtin_amdgcn_wmma_f32_16x16x32_bf16(false,a0,false,b0,(short)0,acc[0][0],false,false);
    acc[0][1] = __builtin_amdgcn_wmma_f32_16x16x32_bf16(false,a0,false,b1,(short)0,acc[0][1],false,false);
    acc[1][0] = __builtin_amdgcn_wmma_f32_16x16x32_bf16(false,a1,false,b0,(short)0,acc[1][0],false,false);
    acc[1][1] = __builtin_amdgcn_wmma_f32_16x16x32_bf16(false,a1,false,b1,(short)0,acc[1][1],false,false);
  }
  // C/D layout: VGPR r: lanes0-15 -> M=r, lanes16-31 -> M=8+r; N = lane&15
#pragma unroll
  for (int im=0; im<2; ++im) {
#pragma unroll
    for (int in=0; in<2; ++in) {
      int col = wn0 + in*16 + nl;
      if (col >= Nout) continue;
      float bv = bias ? bias[col] : 0.f;
#pragma unroll
      for (int rI=0; rI<8; ++rI) {
        int rw = wm0 + im*16 + hf*8 + rI;
        if (rw >= M) continue;
        float v = acc[im][in][rI] + bv;
        if (epi == 1) v = gelu_f(v);
        size_t idx = (size_t)rw*ldc + col;
        if (R) v += R[idx];
        C[idx] = v;
      }
    }
  }
}

// ---------------- small dense linear (per-element dot), batched ------------
__global__ void small_linear(const float* __restrict__ A, ll sA, int lda,
                             const float* __restrict__ Wt, ll sW, int ldw,
                             const float* __restrict__ bias, ll sBias,
                             float* __restrict__ C, ll sC, int ldc,
                             int M, int Nout, int K, int act)
{
  int n = blockIdx.x * blockDim.x + threadIdx.x;
  int m = blockIdx.y; int bb = blockIdx.z;
  if (n >= Nout) return;
  const float* a = A + bb*sA + (size_t)m*lda;
  const float* w = Wt + bb*sW + (size_t)n*ldw;
  float s = bias ? bias[bb*sBias + n] : 0.f;
  for (int kk=0; kk<K; ++kk) s += a[kk]*w[kk];
  C[bb*sC + (size_t)m*ldc + n] = act_apply(s, act);
}

// ---------------- layernorm over D per row ---------------------------------
__global__ __launch_bounds__(256) void layernorm_rows(
    const float* __restrict__ X, ll ldx, const float* __restrict__ w,
    const float* __restrict__ b, float* __restrict__ Y, ll ldy, int D)
{
  int r = blockIdx.x;
  const float* x = X + (size_t)r*ldx;
  float* y = Y + (size_t)r*ldy;
  __shared__ float red[256];
  int tid = threadIdx.x;
  float s = 0.f;
  for (int d=tid; d<D; d+=256) s += x[d];
  red[tid]=s; __syncthreads();
  for (int o=128;o>0;o>>=1){ if(tid<o) red[tid]+=red[tid+o]; __syncthreads(); }
  float mean = red[0]/D; __syncthreads();
  float s2 = 0.f;
  for (int d=tid; d<D; d+=256){ float t=x[d]-mean; s2+=t*t; }
  red[tid]=s2; __syncthreads();
  for (int o=128;o>0;o>>=1){ if(tid<o) red[tid]+=red[tid+o]; __syncthreads(); }
  float inv = rsqrtf(red[0]/D + 1e-5f);
  for (int d=tid; d<D; d+=256) y[d] = (x[d]-mean)*inv*w[d] + b[d];
}

// ---------------- gated softmax over attention row (padded to NPAD) --------
__global__ __launch_bounds__(256) void gated_softmax(
    float* __restrict__ scores, const float* __restrict__ ext,
    const float* __restrict__ lam, const float* __restrict__ th)
{
  int n  = blockIdx.x;          // 0..196
  int bh = blockIdx.y;          // 0..191
  int b = bh / NHEAD, hh = bh % NHEAD;
  float* row = scores + ((size_t)bh*NTOK + n)*NPAD;
  const float* e = ext + (size_t)(b*NHEAD + hh)*NTOK;
  float lamv = lam[b*NHEAD+hh], thv = th[b*NHEAD+hh];
  float en = e[n];
  __shared__ float red[256];
  int tid = threadIdx.x;
  bool have = tid < NTOK;
  float myv = 0.f, v = -INFINITY;
  if (have) {
    float s  = row[tid]*0.125f;                    // SCALE = HD^-0.5
    float gm = en*e[tid];
    float g  = sigmoid_f(lamv*(gm - thv));
    myv = s*g; v = myv;
  }
  red[tid]=v; __syncthreads();
  for (int o=128;o>0;o>>=1){ if(tid<o) red[tid]=fmaxf(red[tid],red[tid+o]); __syncthreads(); }
  float mx = red[0]; __syncthreads();
  float ex = have ? expf(myv-mx) : 0.f;
  red[tid]=ex; __syncthreads();
  for (int o=128;o>0;o>>=1){ if(tid<o) red[tid]+=red[tid+o]; __syncthreads(); }
  float sum = red[0];
  if (have)            row[tid] = ex/sum;
  else if (tid < NPAD) row[tid] = 0.f;   // zero K-padding for context GEMM
}

// ---------------- misc elementwise kernels ---------------------------------
__global__ void patchify_k(const float* __restrict__ x, float* __restrict__ xp)
{
  int idx = blockIdx.x*blockDim.x + threadIdx.x;
  if (idx >= BNP*DIMC) return;
  int ci = idx % DIMC; int p = (idx/DIMC) % NPAT; int b = idx/(DIMC*NPAT);
  int c = ci >> 8; int r = ci & 255; int py = r >> 4; int px = r & 15;
  int gy = p / GRD, gx = p % GRD;
  xp[idx] = x[(((size_t)(b*3 + c)*224) + gy*16+py)*224 + gx*16+px];
}

__global__ void add_cls_pos(float* __restrict__ h, const float* __restrict__ cls,
                            const float* __restrict__ pos)
{
  int idx = blockIdx.x*blockDim.x + threadIdx.x;
  if (idx >= BN*DIMC) return;
  int d = idx % DIMC; int n = (idx/DIMC) % NTOK;
  if (n == 0) h[idx] = cls[d] + pos[d];
  else        h[idx] += pos[(size_t)n*DIMC + d];
}

__global__ void build_ext(const float* __restrict__ pm, const float* __restrict__ ef,
                          const float* __restrict__ mc, const float* __restrict__ nc,
                          const float* __restrict__ wall, float* __restrict__ ext)
{
  int idx = blockIdx.x*blockDim.x + threadIdx.x;
  if (idx >= BATCH*NHEAD*NTOK) return;
  int q1 = idx % NTOK; int t = idx/NTOK; int ch = t % NHEAD; int b = t / NHEAD;
  float v = 0.f;
  if (q1 > 0) {
    int q = q1 - 1;
    const float* P = pm + (size_t)b*8*NPAT + q;
    #define PMv(i) P[(i)*NPAT]
    if      (ch < 8)  v = PMv(ch);
    else if (ch == 8) v = ef[0]*PMv(0)+ef[1]*PMv(1)+ef[2]*PMv(7);
    else if (ch == 9) v = mc[0]*PMv(3)+mc[1]*PMv(4)+mc[2]*PMv(5);
    else if (ch ==10) v = nc[0]*PMv(2)+nc[1]*PMv(6);
    else { for (int i=0;i<8;++i) v += wall[i]*PMv(i); }
    #undef PMv
  }
  ext[idx] = v;
}

__global__ void token_mean_max(const float* __restrict__ X, int Ntok, int D,
                               float* __restrict__ outMean, int ldm,
                               float* __restrict__ outMax, int ldx2)
{
  int idx = blockIdx.x*blockDim.x + threadIdx.x;
  if (idx >= BATCH*D) return;
  int b = idx / D, d = idx % D;
  const float* p = X + (size_t)b*Ntok*D + d;
  float s = 0.f, mx = -INFINITY;
  for (int n=0;n<Ntok;++n){ float v = p[(size_t)n*D]; s += v; mx = fmaxf(mx, v); }
  outMean[(size_t)b*ldm + d] = s/(float)Ntok;
  if (outMax) outMax[(size_t)b*ldx2 + d] = mx;
}

__global__ void concat_rows(float* __restrict__ dst, int ldd,
                            const float* __restrict__ s1, int ld1, int w1,
                            const float* __restrict__ s2, int ld2, int w2, int rows)
{
  int idx = blockIdx.x*blockDim.x + threadIdx.x;
  int w = w1 + w2;
  if (idx >= rows*w) return;
  int r = idx / w, c = idx % w;
  dst[(size_t)r*ldd + c] = (c < w1) ? s1[(size_t)r*ld1 + c] : s2[(size_t)r*ld2 + (c-w1)];
}

__global__ void softmax_small(const float* __restrict__ X, float* __restrict__ Y,
                              int rows, int w)
{
  int r = blockIdx.x*blockDim.x + threadIdx.x;
  if (r >= rows) return;
  const float* x = X + (size_t)r*w; float* y = Y + (size_t)r*w;
  float mx = -INFINITY; for (int i=0;i<w;++i) mx = fmaxf(mx, x[i]);
  float s = 0.f; for (int i=0;i<w;++i){ float e = expf(x[i]-mx); y[i]=e; s+=e; }
  for (int i=0;i<w;++i) y[i] /= s;
}

__global__ void mw_combine(const float* __restrict__ gsig, const float* __restrict__ bsig,
                           const float* __restrict__ cur, const float* __restrict__ prev,
                           float* __restrict__ mw, float* __restrict__ outp)
{
  int idx = threadIdx.x;      // 0..191
  if (idx >= BATCH*NHEAD) return;
  int b = idx / NHEAD;
  float v = gsig[b]*cur[idx] + bsig[b]*prev[idx];
  mw[idx] = v;
  if (outp) outp[idx] = v;
}

__global__ void add_rowvec(float* __restrict__ dst, const float* __restrict__ src,
                           const float* __restrict__ rv, int rows, int w)
{
  int idx = blockIdx.x*blockDim.x + threadIdx.x;
  if (idx >= rows*w) return;
  dst[idx] = src[idx] + rv[idx % w];
}

__global__ void fill_k(float* __restrict__ p, float v, int n)
{
  int idx = blockIdx.x*blockDim.x + threadIdx.x;
  if (idx < n) p[idx] = v;
}

// split qkv and build V^T with zero padding on the token axis (NPAD)
__global__ void split_qkv(const float* __restrict__ qkv, float* __restrict__ q,
                          float* __restrict__ k, float* __restrict__ vt)
{
  int idx = blockIdx.x*blockDim.x + threadIdx.x;
  if (idx >= BATCH*NHEAD*NPAD*HDIM) return;
  int d = idx & 63; int t = idx >> 6; int n = t % NPAD; t /= NPAD;
  int hh = t % NHEAD; int b = t / NHEAD;
  bool valid = n < NTOK;
  size_t src = ((size_t)(b*NTOK + (valid ? n : 0)))*(3*DIMC) + hh*HDIM + d;
  float vv = valid ? qkv[src + 2*DIMC] : 0.f;
  vt[(((size_t)(b*NHEAD + hh))*HDIM + d)*NPAD + n] = vv;
  if (valid) {
    size_t dst = (((size_t)(b*NHEAD + hh))*NTOK + n)*HDIM + d;
    q[dst] = qkv[src];
    k[dst] = qkv[src + DIMC];
  }
}

__global__ void scale_mw(float* __restrict__ x, const float* __restrict__ mw)
{
  int idx = blockIdx.x*blockDim.x + threadIdx.x;
  if (idx >= BN*DIMC) return;
  int d = idx % DIMC; int b = (idx/DIMC)/NTOK;
  x[idx] *= mw[b*NHEAD + d/HDIM];
}

__global__ void resid3(float* __restrict__ h, const float* __restrict__ ao,
                       const float* __restrict__ hp)
{
  int idx = blockIdx.x*blockDim.x + threadIdx.x;
  if (idx >= BN*DIMC) return;
  int d = idx % DIMC; int b = (idx/DIMC)/NTOK;
  h[idx] += ao[idx] + hp[(size_t)b*DIMC + d];
}

__global__ __launch_bounds__(768) void head_attn(const float* __restrict__ hqv,
                                                 float* __restrict__ inter)
{
  int b = blockIdx.x;
  __shared__ float scm[NHEAD][NHEAD];
  __shared__ float aw[NHEAD][NHEAD];
  int tid = threadIdx.x;
  if (tid < NHEAD*NHEAD) {
    int hh = tid / NHEAD, g = tid % NHEAD;
    const float* qr = hqv + (size_t)(b*NHEAD + hh)*192;
    const float* kr = hqv + (size_t)(b*NHEAD + g )*192 + 64;
    float s = 0.f;
    for (int d=0; d<HDIM; ++d) s += qr[d]*kr[d];
    scm[hh][g] = s*0.125f;
  }
  __syncthreads();
  if (tid < NHEAD) {
    float mx = -INFINITY;
    for (int g=0; g<NHEAD; ++g) mx = fmaxf(mx, scm[tid][g]);
    float sum = 0.f;
    for (int g=0; g<NHEAD; ++g){ float e = expf(scm[tid][g]-mx); aw[tid][g]=e; sum+=e; }
    for (int g=0; g<NHEAD; ++g) aw[tid][g] /= sum;
  }
  __syncthreads();
  int hh = tid >> 6, d = tid & 63;
  float s = 0.f;
  for (int g=0; g<NHEAD; ++g) s += aw[hh][g]*hqv[(size_t)(b*NHEAD + g)*192 + 128 + d];
  inter[(size_t)b*DIMC + tid] = s;
}

// ---------------------------------------------------------------------------
// Input index map (setup_inputs insertion order, nested dicts in insertion
// order, _lin inserts 'w' then 'b').
// ---------------------------------------------------------------------------
enum {
  I_X=0, I_RMASK, I_PATCH_W, I_PATCH_B, I_CLS, I_POS, I_MPW, I_MPB,
  I_CW_EF, I_CW_MC, I_CW_NC, I_CW_ALL, I_LAYER_EMB,
  I_GFF_W, I_GFF_B, I_CTXFC_W, I_CTXFC_B, I_CTXLN_W, I_CTXLN_B,
  I_RI1_W, I_RI1_B, I_RI2_W, I_RI2_B, I_MEM_W, I_MEM_B, I_NEW_W, I_NEW_B,
  I_LAM1_W, I_LAM1_B, I_LAM2_W, I_LAM2_B, I_TH1_W, I_TH1_B, I_TH2_W, I_TH2_B,
  I_BASE_TH, I_BLK0
};
enum { BO_N1W=0, BO_N1B, BO_QKVW, BO_QKVB, BO_APW, BO_APB, BO_HQW, BO_HQB,
       BO_HPW, BO_HPB, BO_N2W, BO_N2B, BO_FC1W, BO_FC1B, BO_FC2W, BO_FC2B };
#define I_NORM_W (I_BLK0 + 16*NLAYER)
#define I_NORM_B (I_NORM_W + 1)
#define I_HEAD_W (I_NORM_W + 2)
#define I_HEAD_B (I_NORM_W + 3)

static inline void gemm(hipStream_t s,
                        const float* A, ll sA1, ll sA2,
                        const float* Wt, ll sW1, ll sW2,
                        const float* bias, float* C, ll sC1, ll sC2,
                        const float* R, int M, int Nout, int K,
                        int lda, int ldw, int ldc, int nb2, int nb, int epi)
{
  dim3 g((Nout+63)/64, (M+63)/64, nb);
  wmma_gemm_nt<<<g, 128, 0, s>>>(A,sA1,sA2,Wt,sW1,sW2,bias,C,sC1,sC2,R,
                                 M,Nout,K,lda,ldw,ldc,nb2,epi);
}

static inline void slin(hipStream_t s,
                        const float* A, ll sA, int lda,
                        const float* Wt, ll sW, int ldw,
                        const float* bias, ll sBias,
                        float* C, ll sC, int ldc,
                        int M, int Nout, int K, int nb, int act)
{
  dim3 g((Nout+63)/64, M, nb);
  small_linear<<<g, 64, 0, s>>>(A,sA,lda,Wt,sW,ldw,bias,sBias,C,sC,ldc,M,Nout,K,act);
}

#define EWL(total) <<<(((total)+255)/256), 256, 0, stream>>>

extern "C" void kernel_launch(void* const* d_in, const int* in_sizes, int n_in,
                              void* d_out, int out_size, void* d_ws, size_t ws_size,
                              hipStream_t stream)
{
  (void)in_sizes; (void)n_in; (void)out_size; (void)ws_size;
  const float* const* F = (const float* const*)d_in;
  float* out = (float*)d_out;

  // -------- workspace bump allocator (floats) ------------------------------
  float* Wb = (float*)d_ws; size_t off = 0;
  auto alloc = [&](size_t n)->float* { float* p = Wb + off; off += n; return p; };
  float* h     = alloc((size_t)BN*DIMC);
  float* hn    = alloc((size_t)BN*DIMC);
  float* big   = alloc((size_t)BN*MLPD);              // xp / qkv-out / mlp1
  float* q     = alloc((size_t)BN*DIMC);
  float* k     = alloc((size_t)BN*DIMC);
  float* vt    = alloc((size_t)BATCH*NHEAD*HDIM*NPAD); // padded token axis
  float* sc    = alloc((size_t)BATCH*NHEAD*NTOK*NPAD); // padded row stride
  float* aopre = alloc((size_t)BN*DIMC);
  float* ao    = alloc((size_t)BN*DIMC);
  float* pm    = alloc(BATCH*8*NPAT);
  float* ext   = alloc(BATCH*NHEAD*NTOK);
  float* wall  = alloc(8);
  float* hstat = alloc(BATCH*2*DIMC);
  float* gf    = alloc(BATCH*DIMC);
  float* ctxin = alloc(BATCH*(DIMC+64));
  float* ctxg  = alloc(BATCH*DIMC);
  float* ctx   = alloc(BATCH*DIMC);
  float* riin  = alloc(BATCH*(DIMC+NHEAD));
  float* t64   = alloc(BATCH*64);
  float* t12   = alloc(BATCH*NHEAD);
  float* cur   = alloc(BATCH*NHEAD);
  float* bsig  = alloc(BATCH);
  float* gsig  = alloc(BATCH);
  float* mw0   = alloc(BATCH*NHEAD);
  float* mw1   = alloc(BATCH*NHEAD);
  float* pibuf = alloc(BATCH*(DIMC+NHEAD));
  float* t32   = alloc(BATCH*32);
  float* lamb  = alloc(BATCH*NHEAD);
  float* thtmp = alloc(BATCH*NHEAD);
  float* thb   = alloc(BATCH*NHEAD);
  float* gmean = alloc(BATCH*DIMC);
  float* hqv   = alloc(192*192);
  float* inter = alloc(BATCH*DIMC);
  float* hp    = alloc(BATCH*DIMC);
  float* clsf  = alloc(BATCH*DIMC);

  // -------- patch embed + positional --------------------------------------
  patchify_k EWL(BNP*DIMC) (F[I_X], big);
  gemm(stream, big, (ll)NPAT*DIMC, 0, F[I_PATCH_W], 0, 0, F[I_PATCH_B],
       h + DIMC, (ll)NTOK*DIMC, 0, nullptr,
       NPAT, DIMC, DIMC, DIMC, DIMC, DIMC, 1, BATCH, 0);
  add_cls_pos EWL(BN*DIMC) (h, F[I_CLS], F[I_POS]);

  // -------- mask pipeline --------------------------------------------------
  slin(stream, F[I_RMASK], NPAT, 8*NPAT, F[I_MPW], (ll)NPAT*NPAT, NPAT,
       F[I_MPB], NPAT, pm, NPAT, 8*NPAT, BATCH, NPAT, NPAT, 8, 0);
  softmax_small<<<1, 32, 0, stream>>>(F[I_CW_ALL], wall, 1, 8);
  build_ext EWL(BATCH*NHEAD*NTOK) (pm, F[I_CW_EF], F[I_CW_MC], F[I_CW_NC], wall, ext);

  // -------- lamm init ------------------------------------------------------
  fill_k EWL(BATCH*NHEAD) (mw0, 1.0f/(float)NHEAD, BATCH*NHEAD);
  float* prevp = mw0;
  float* mwp   = mw1;

  for (int li = 0; li < NLAYER; ++li) {
    const float* const* Bp = F + I_BLK0 + 16*li;
    const float* le = F[I_LAYER_EMB] + li*64;

    // ---- lamm(h, li, prev) -> mw, lam, th --------------------------------
    token_mean_max EWL(BATCH*DIMC) (h, NTOK, DIMC, hstat, 2*DIMC, hstat + DIMC, 2*DIMC);
    slin(stream, hstat, 0, 2*DIMC, F[I_GFF_W], 0, 2*DIMC, F[I_GFF_B], 0,
         gf, 0, DIMC, BATCH, DIMC, 2*DIMC, 1, 0);
    concat_rows EWL(BATCH*(DIMC+64)) (ctxin, DIMC+64, gf, DIMC, DIMC, le, 0, 64, BATCH);
    slin(stream, ctxin, 0, DIMC+64, F[I_CTXFC_W], 0, DIMC+64, F[I_CTXFC_B], 0,
         ctxg, 0, DIMC, BATCH, DIMC, DIMC+64, 1, 1);
    layernorm_rows<<<BATCH, 256, 0, stream>>>(ctxg, DIMC, F[I_CTXLN_W], F[I_CTXLN_B], ctx, DIMC, DIMC);
    concat_rows EWL(BATCH*(DIMC+NHEAD)) (riin, DIMC+NHEAD, ctx, DIMC, DIMC, prevp, NHEAD, NHEAD, BATCH);
    slin(stream, riin, 0, DIMC+NHEAD, F[I_RI1_W], 0, DIMC+NHEAD, F[I_RI1_B], 0,
         t64, 0, 64, BATCH, 64, DIMC+NHEAD, 1, 1);
    slin(stream, t64, 0, 64, F[I_RI2_W], 0, 64, F[I_RI2_B], 0,
         t12, 0, NHEAD, BATCH, NHEAD, 64, 1, 0);
    softmax_small<<<1, 32, 0, stream>>>(t12, cur, BATCH, NHEAD);
    slin(stream, ctx, 0, DIMC, F[I_MEM_W], 0, DIMC, F[I_MEM_B], 0, bsig, 0, 1, BATCH, 1, DIMC, 1, 3);
    slin(stream, ctx, 0, DIMC, F[I_NEW_W], 0, DIMC, F[I_NEW_B], 0, gsig, 0, 1, BATCH, 1, DIMC, 1, 3);
    float* mwout = (li >= NLAYER-3) ? (out + 32 + (li-(NLAYER-3))*BATCH*NHEAD) : nullptr;
    mw_combine<<<1, 192, 0, stream>>>(gsig, bsig, cur, prevp, mwp, mwout);
    concat_rows EWL(BATCH*(DIMC+NHEAD)) (pibuf, DIMC+NHEAD, ctx, DIMC, DIMC, mwp, NHEAD, NHEAD, BATCH);
    slin(stream, pibuf, 0, DIMC+NHEAD, F[I_LAM1_W], 0, DIMC+NHEAD, F[I_LAM1_B], 0,
         t32, 0, 32, BATCH, 32, DIMC+NHEAD, 1, 2);
    slin(stream, t32, 0, 32, F[I_LAM2_W], 0, 32, F[I_LAM2_B], 0,
         lamb, 0, NHEAD, BATCH, NHEAD, 32, 1, 4);
    slin(stream, pibuf, 0, DIMC+NHEAD, F[I_TH1_W], 0, DIMC+NHEAD, F[I_TH1_B], 0,
         t32, 0, 32, BATCH, 32, DIMC+NHEAD, 1, 2);
    slin(stream, t32, 0, 32, F[I_TH2_W], 0, 32, F[I_TH2_B], 0,
         thtmp, 0, NHEAD, BATCH, NHEAD, 32, 1, 2);
    add_rowvec EWL(BATCH*NHEAD) (thb, thtmp, F[I_BASE_TH], BATCH, NHEAD);

    // ---- rgmha ------------------------------------------------------------
    layernorm_rows<<<BN, 256, 0, stream>>>(h, DIMC, Bp[BO_N1W], Bp[BO_N1B], hn, DIMC, DIMC);
    gemm(stream, hn, 0, 0, Bp[BO_QKVW], 0, 0, Bp[BO_QKVB], big, 0, 0, nullptr,
         BN, 3*DIMC, DIMC, DIMC, DIMC, 3*DIMC, 1, 1, 0);
    split_qkv EWL(BATCH*NHEAD*NPAD*HDIM) (big, q, k, vt);
    // scores[bh] = q[bh] @ k[bh]^T   (ldc padded to NPAD)
    gemm(stream, q, (ll)NTOK*HDIM, 0, k, (ll)NTOK*HDIM, 0, nullptr,
         sc, (ll)NTOK*NPAD, 0, nullptr,
         NTOK, NTOK, HDIM, HDIM, HDIM, NPAD, 1, BATCH*NHEAD, 0);
    gated_softmax<<<dim3(NTOK, BATCH*NHEAD), 256, 0, stream>>>(sc, ext, lamb, thb);
    // ao_pre[b,n,h*64+d] = attn[bh] @ v[bh]  (K = NPAD, zero padded)
    gemm(stream, sc, (ll)NHEAD*NTOK*NPAD, (ll)NTOK*NPAD,
         vt, (ll)NHEAD*HDIM*NPAD, (ll)HDIM*NPAD, nullptr,
         aopre, (ll)NTOK*DIMC, (ll)HDIM, nullptr,
         NTOK, HDIM, NPAD, NPAD, NPAD, DIMC, NHEAD, BATCH*NHEAD, 0);
    scale_mw EWL(BN*DIMC) (aopre, mwp);
    gemm(stream, aopre, 0, 0, Bp[BO_APW], 0, 0, Bp[BO_APB], ao, 0, 0, nullptr,
         BN, DIMC, DIMC, DIMC, DIMC, DIMC, 1, 1, 0);

    // ---- head_inter -------------------------------------------------------
    token_mean_max EWL(BATCH*DIMC) (ao, NTOK, DIMC, gmean, DIMC, nullptr, 0);
    slin(stream, gmean, 0, HDIM, Bp[BO_HQW], 0, HDIM, Bp[BO_HQB], 0,
         hqv, 0, 192, BATCH*NHEAD, 192, HDIM, 1, 0);
    head_attn<<<BATCH, 768, 0, stream>>>(hqv, inter);
    slin(stream, inter, 0, DIMC, Bp[BO_HPW], 0, DIMC, Bp[BO_HPB], 0,
         hp, 0, DIMC, BATCH, DIMC, DIMC, 1, 0);
    resid3 EWL(BN*DIMC) (h, ao, hp);

    // ---- MLP --------------------------------------------------------------
    layernorm_rows<<<BN, 256, 0, stream>>>(h, DIMC, Bp[BO_N2W], Bp[BO_N2B], hn, DIMC, DIMC);
    gemm(stream, hn, 0, 0, Bp[BO_FC1W], 0, 0, Bp[BO_FC1B], big, 0, 0, nullptr,
         BN, MLPD, DIMC, DIMC, DIMC, MLPD, 1, 1, 1);                  // +GELU
    gemm(stream, big, 0, 0, Bp[BO_FC2W], 0, 0, Bp[BO_FC2B], h, 0, 0, h,
         BN, DIMC, MLPD, MLPD, MLPD, DIMC, 1, 1, 0);                  // +resid

    float* tswap = prevp; prevp = mwp; mwp = tswap;
  }

  // -------- final norm (cls rows only) + head ------------------------------
  layernorm_rows<<<BATCH, 256, 0, stream>>>(h, (ll)NTOK*DIMC, F[I_NORM_W], F[I_NORM_B],
                                            clsf, DIMC, DIMC);
  slin(stream, clsf, 0, DIMC, F[I_HEAD_W], 0, DIMC, F[I_HEAD_B], 0,
       out, 0, 2, BATCH, 2, DIMC, 1, 0);
}
